// FaustModel_65695819759771
// MI455X (gfx1250) — compile-verified, matrix operand order checked by reading, not run on hardware
//
#include <hip/hip_runtime.h>
#include <hip/hip_bf16.h>

// ---------------------------------------------------------------------------
// FAUST mesh-CNN inference for MI455X (gfx1250, wave32, WMMA).
// Heavy math: bf16 WMMA (v_wmma_f32_16x16x32_bf16) with f32 accumulation.
// Conv kernel: 8 rotation accumulators per t-subtile; B tiles staged through
// LDS with async global->LDS copies (double-buffered, ASYNCcnt-tracked).
// ---------------------------------------------------------------------------

typedef __attribute__((ext_vector_type(16))) __bf16 v16bf;
typedef __attribute__((ext_vector_type(8)))  __bf16 v8bf;
typedef __attribute__((ext_vector_type(8)))  float  v8f;
typedef __attribute__((ext_vector_type(4)))  int    v4i;

typedef __attribute__((address_space(1))) v4i gv4i;   // global int4
typedef __attribute__((address_space(3))) v4i lv4i;   // LDS int4

#define N_VERTS 6890
#define SIG_DIM 544
#define DOWN    64

#if defined(__has_builtin)
#if __has_builtin(__builtin_amdgcn_global_load_async_to_lds_b128)
#define HAVE_ASYNC_LDS 1
#endif
#if __has_builtin(__builtin_amdgcn_s_wait_asynccnt)
#define HAVE_WAIT_ASYNC 1
#endif
#endif

// 16-byte async global->LDS copy (falls back to a synchronous LDS copy).
static __device__ __forceinline__ void async_cp16(const __bf16* g, __bf16* l) {
#ifdef HAVE_ASYNC_LDS
  __builtin_amdgcn_global_load_async_to_lds_b128(
      (gv4i*)(void*)g, (lv4i*)(void*)l, 0, 0);
#else
  *(v8bf*)l = *(const v8bf*)g;
#endif
}

template <int N>
static __device__ __forceinline__ void wait_async() {
#ifdef HAVE_ASYNC_LDS
#ifdef HAVE_WAIT_ASYNC
  __builtin_amdgcn_s_wait_asynccnt(N);
#else
  asm volatile("s_wait_asynccnt %0" ::"i"(N) : "memory");
#endif
#endif
}

// Per-lane WMMA 16-bit operand fragment: two contiguous 16-byte chunks
// (K offsets [0..7] and [16..23]; the lane's 0/8 half-select is baked into
// the base pointer). Matches the ISA 16-bit 16x32 A/B VGPR layout.
static __device__ __forceinline__ v16bf load_frag16(const __bf16* p) {
  v8bf lo = *(const v8bf*)(p);
  v8bf hi = *(const v8bf*)(p + 16);
  return __builtin_shufflevector(lo, hi, 0,1,2,3,4,5,6,7,8,9,10,11,12,13,14,15);
}

// ---------------------------------------------------------------------------
// Stage 1: normalization + Dense(544->64) + relu + BN scale/shift.
// ---------------------------------------------------------------------------
__global__ __launch_bounds__(64) void downsize_kernel(
    const float* __restrict__ sig, const float* __restrict__ mean,
    const float* __restrict__ var, const float* __restrict__ wd,
    const float* __restrict__ bd,  const float* __restrict__ bs,
    const float* __restrict__ bb,  float* __restrict__ x0)
{
  __shared__ float row[SIG_DIM];
  const int n = blockIdx.x;
  const int j = threadIdx.x;
  for (int c = j; c < SIG_DIM; c += 64)
    row[c] = (sig[(size_t)n * SIG_DIM + c] - mean[c]) * rsqrtf(var[c]);
  __syncthreads();
  float acc = bd[j];
  for (int c = 0; c < SIG_DIM; ++c)
    acc = fmaf(row[c], wd[c * DOWN + j], acc);   // wd is (544,64): coalesced in j
  acc = fmaxf(acc, 0.0f);
  x0[(size_t)n * DOWN + j] = acc * bs[j] + bb[j];
}

// ---------------------------------------------------------------------------
// Barycentric patch gather: A[nLoc, (r*8+a)*C + c] = sum_k x[idx]*w, as bf16.
// ---------------------------------------------------------------------------
__global__ __launch_bounds__(256) void interp_gather(
    const float* __restrict__ x,      // N x C (full signal of this layer)
    const int*   __restrict__ idx,    // N*40*3
    const float* __restrict__ wgt,    // N*40*3
    __bf16* __restrict__ A,           // rows x (40*C), chunk local
    int rowStart, long long total, int C)
{
  long long gid = (long long)blockIdx.x * blockDim.x + threadIdx.x;
  if (gid >= total) return;
  int c = (int)(gid % C);
  long long ns = gid / C;                          // local n*40 + seg
  long long nsg = ns + (long long)rowStart * 40;
  const int*   ip = idx + nsg * 3;
  const float* wp = wgt + nsg * 3;
  float v = x[(size_t)ip[0] * C + c] * wp[0]
          + x[(size_t)ip[1] * C + c] * wp[1]
          + x[(size_t)ip[2] * C + c] * wp[2];
  A[gid] = (__bf16)v;
}

// ---------------------------------------------------------------------------
// fp32 -> bf16 elementwise convert.
// ---------------------------------------------------------------------------
__global__ __launch_bounds__(256) void cvt_f32_bf16(
    const float* __restrict__ src, __bf16* __restrict__ dst, long long n)
{
  long long i = (long long)blockIdx.x * blockDim.x + threadIdx.x;
  if (i < n) dst[i] = (__bf16)src[i];
}

// w: K x Ncols (row-major) -> bt: Ncols x K bf16 (row-major).
__global__ __launch_bounds__(256) void transpose_cvt(
    const float* __restrict__ w, __bf16* __restrict__ bt,
    int K, int Ncols, long long total)
{
  long long i = (long long)blockIdx.x * blockDim.x + threadIdx.x;
  if (i >= total) return;
  int col = (int)(i % Ncols);
  int k   = (int)(i / Ncols);
  bt[(size_t)col * K + k] = (__bf16)w[i];
}

// ---------------------------------------------------------------------------
// Fused ConvDirac (all 8 rotations) + AngularMaxPool + bias/relu/BN.
// Wave tile: 16 rows (n) x 32 cols (t); 8 rotation accumulators per subtile.
//   out[rot] += A[a] (x) B[(a+rot) & 7]
// B tile per (radial, 32-chan chunk) group = 8 segs x 32 t x 32 ch bf16,
// staged once per workgroup into LDS via async copies, double-buffered.
// LDS rows padded to 40 bf16 (80B stride -> conflict-free 16B ds loads).
// ---------------------------------------------------------------------------
#define BROW 40                      // padded LDS row (32 ch + 8 pad)
#define BTILE (8 * 32 * BROW)        // one B tile in LDS (bf16 elements)

__global__ __launch_bounds__(256) void isc_conv_wmma(
    const __bf16* __restrict__ A,    // rows x K (chunk), K = 40*C
    const __bf16* __restrict__ Bk,   // T x K  (kernel bf16, k = (r,a,c))
    const float*  __restrict__ bias,
    const float*  __restrict__ bn_s,
    const float*  __restrict__ bn_b,
    float* __restrict__ Xout,        // N x T (global rows)
    int rows, int rowStart, int T, int C)
{
  __shared__ __bf16 ldsB[2][BTILE];  // 2 x 20 KB

  const int lane = threadIdx.x & 31;
  const int wave = threadIdx.x >> 5;
  const int tid  = threadIdx.x;
  const int mBase = blockIdx.y * 128 + wave * 16;   // wave-uniform
  const int tBase = blockIdx.x * 32;

  const int K = 40 * C;
  int mLoc = mBase + (lane & 15);
  if (mLoc >= rows) mLoc = rows - 1;                // clamp; store is guarded
  const int kHalf = (lane < 16) ? 0 : 8;
  const __bf16* aRow = A + (size_t)mLoc * K + kHalf;

  // Issue the async B-tile copy for segment base `seg` into buffer `buf`.
  // 1024 x 16B transfers; 4 per thread.
  auto issueB = [&](int seg, int buf) {
#pragma unroll
    for (int i = 0; i < 4; ++i) {
      const int t = tid + (i << 8);                 // 0..1023
      const int q  = t & 3;                         // 16B quarter of 64B row
      const int tl = (t >> 2) & 31;                 // t-col within tile
      const int ap = t >> 7;                        // angular segment 0..7
      const __bf16* g = Bk + (size_t)(tBase + tl) * K + seg + ap * C + q * 8;
      __bf16* l = &ldsB[buf][(ap * 32 + tl) * BROW + q * 8];
      async_cp16(g, l);
    }
  };

  v8f acc0[8], acc1[8];
#pragma unroll
  for (int i = 0; i < 8; ++i) { acc0[i] = (v8f){}; acc1[i] = (v8f){}; }

  const int tl0 = lane & 15;
  int buf = 0;
  issueB(0, 0);                                     // prologue: group 0

  for (int r = 0; r < 5; ++r) {
    for (int c0 = 0; c0 < C; c0 += 32) {
      const int segBase = (r * 8) * C + c0;
      int nextSeg = -1;
      if (c0 + 32 < C)      nextSeg = segBase + 32;
      else if (r + 1 < 5)   nextSeg = (r + 1) * 8 * C;

      if (nextSeg >= 0) { issueB(nextSeg, buf ^ 1); wait_async<4>(); }
      else              {                            wait_async<0>(); }
      __syncthreads();                              // B tile visible WG-wide

      // Prefetch next group's A data (speculative; OOB is dropped by HW).
      __builtin_prefetch(aRow + segBase + 8 * C, 0, 1);

      v16bf af[8];
#pragma unroll
      for (int a = 0; a < 8; ++a)
        af[a] = load_frag16(aRow + segBase + a * C);

#pragma unroll
      for (int ap = 0; ap < 8; ++ap) {              // B angular index
        const __bf16* bb = &ldsB[buf][(ap * 32) * BROW + kHalf];
        v16bf b0 = load_frag16(bb + tl0 * BROW);
        v16bf b1 = load_frag16(bb + (tl0 + 16) * BROW);
#pragma unroll
        for (int rot = 0; rot < 8; ++rot) {
          const int a = (ap - rot) & 7;             // A angular partner
          acc0[rot] = __builtin_amdgcn_wmma_f32_16x16x32_bf16(
              false, af[a], false, b0, (short)0, acc0[rot], false, false);
          acc1[rot] = __builtin_amdgcn_wmma_f32_16x16x32_bf16(
              false, af[a], false, b1, (short)0, acc1[rot], false, false);
        }
      }
      __syncthreads();                              // done reading this buf
      buf ^= 1;
    }
  }

  // Angular max-pool over rotations (relu∘max == max∘relu).
  v8f pool0 = acc0[0], pool1 = acc1[0];
#pragma unroll
  for (int rot = 1; rot < 8; ++rot) {
#pragma unroll
    for (int i = 0; i < 8; ++i) {
      pool0[i] = fmaxf(pool0[i], acc0[rot][i]);
      pool1[i] = fmaxf(pool1[i], acc1[rot][i]);
    }
  }

  // Epilogue: relu(max_rot + bias) * bn_s + bn_b.  D layout: VGPR i holds
  // (M = i, N = lane) for lanes 0-15 and (M = i+8, N = lane-16) for 16-31.
  const int colA = tBase + (lane & 15);
  const int colB = colA + 16;                       // T % 32 == 0 -> in range
  const float biasA = bias[colA], sA = bn_s[colA], bA = bn_b[colA];
  const float biasB = bias[colB], sB = bn_s[colB], bB = bn_b[colB];
  const int rOff = (lane >> 4) << 3;
#pragma unroll
  for (int i = 0; i < 8; ++i) {
    const int rl = mBase + rOff + i;
    if (rl < rows) {
      const size_t rg = (size_t)(rowStart + rl);
      Xout[rg * T + colA] = fmaxf(pool0[i] + biasA, 0.0f) * sA + bA;
      Xout[rg * T + colB] = fmaxf(pool1[i] + biasB, 0.0f) * sB + bB;
    }
  }
}

// ---------------------------------------------------------------------------
// Final Dense: Out[m, n] = x[m, :] . w_out[:, n] + b_out[n].
// A: M x K bf16, Bt: Ncols x K bf16 (w_out transposed). Edge-guarded.
// ---------------------------------------------------------------------------
__global__ __launch_bounds__(256) void gemm_out_wmma(
    const __bf16* __restrict__ A,    // M x K
    const __bf16* __restrict__ Bt,   // Ncols x K
    const float*  __restrict__ bias, // Ncols
    float* __restrict__ Out,         // M x Ncols
    int M, int Ncols, int K)
{
  const int lane = threadIdx.x & 31;
  const int wave = threadIdx.x >> 5;
  const int mBase = blockIdx.y * 128 + wave * 16;
  const int tBase = blockIdx.x * 32;
  if (mBase >= M) return;

  int mLoc = mBase + (lane & 15);
  if (mLoc >= M) mLoc = M - 1;
  const __bf16* aRow = A + (size_t)mLoc * K;
  const int kHalf = (lane < 16) ? 0 : 8;
  const int t0 = tBase + (lane & 15);
  const int t1 = t0 + 16;
  const int t0c = (t0 < Ncols) ? t0 : (Ncols - 1);
  const int t1c = (t1 < Ncols) ? t1 : (Ncols - 1);
  const __bf16* bRow0 = Bt + (size_t)t0c * K;
  const __bf16* bRow1 = Bt + (size_t)t1c * K;

  v8f acc0 = {}, acc1 = {};
  for (int k0 = 0; k0 < K; k0 += 32) {
    v16bf af = load_frag16(aRow + k0 + kHalf);
    v16bf b0 = load_frag16(bRow0 + k0 + kHalf);
    v16bf b1 = load_frag16(bRow1 + k0 + kHalf);
    acc0 = __builtin_amdgcn_wmma_f32_16x16x32_bf16(false, af, false, b0,
                                                   (short)0, acc0, false, false);
    acc1 = __builtin_amdgcn_wmma_f32_16x16x32_bf16(false, af, false, b1,
                                                   (short)0, acc1, false, false);
  }

  const float biasA = (t0 < Ncols) ? bias[t0] : 0.0f;
  const float biasB = (t1 < Ncols) ? bias[t1] : 0.0f;
  const int rOff = (lane >> 4) << 3;
#pragma unroll
  for (int i = 0; i < 8; ++i) {
    const int rr = mBase + rOff + i;
    if (rr < M) {
      const size_t o = (size_t)rr * Ncols;
      if (t0 < Ncols) Out[o + t0] = acc0[i] + biasA;
      if (t1 < Ncols) Out[o + t1] = acc1[i] + biasB;
    }
  }
}

// ---------------------------------------------------------------------------
// Host-side orchestration (graph-capture safe: only kernel launches).
// ---------------------------------------------------------------------------
extern "C" void kernel_launch(void* const* d_in, const int* in_sizes, int n_in,
                              void* d_out, int out_size, void* d_ws, size_t ws_size,
                              hipStream_t stream)
{
  (void)in_sizes; (void)n_in; (void)out_size;
  const float* signal = (const float*)d_in[0];
  const int*   bc_idx = (const int*)d_in[1];
  const float* bc_w   = (const float*)d_in[2];
  const float* nmean  = (const float*)d_in[3];
  const float* nvar   = (const float*)d_in[4];
  const float* w_down = (const float*)d_in[5];
  const float* b_down = (const float*)d_in[6];
  const float* bnd_s  = (const float*)d_in[7];
  const float* bnd_b  = (const float*)d_in[8];
  const float* kerns[4] = {(const float*)d_in[9],  (const float*)d_in[13],
                           (const float*)d_in[17], (const float*)d_in[21]};
  const float* cbias[4] = {(const float*)d_in[10], (const float*)d_in[14],
                           (const float*)d_in[18], (const float*)d_in[22]};
  const float* bns[4]   = {(const float*)d_in[11], (const float*)d_in[15],
                           (const float*)d_in[19], (const float*)d_in[23]};
  const float* bnb[4]   = {(const float*)d_in[12], (const float*)d_in[16],
                           (const float*)d_in[20], (const float*)d_in[24]};
  const float* w_out = (const float*)d_in[25];
  const float* b_out = (const float*)d_in[26];
  float* out = (float*)d_out;

  const int outDims[4] = {96, 256, 384, 384};
  const int N = N_VERTS;

  // Workspace carve-out (256B aligned).
  char* wsp = (char*)d_ws;
  auto carve = [&](size_t bytes) -> char* {
    char* p = wsp; wsp += (bytes + 255) & ~(size_t)255; return p;
  };
  __bf16* Bk = (__bf16*)carve((size_t)384 * 15360 * sizeof(__bf16)); // max T*K
  float*  x0 = (float*) carve((size_t)N * 384 * sizeof(float));
  float*  x1 = (float*) carve((size_t)N * 384 * sizeof(float));
  __bf16* xa = (__bf16*)carve((size_t)N * 384 * sizeof(__bf16));
  __bf16* Bt = (__bf16*)carve((size_t)N * 384 * sizeof(__bf16));
  const size_t used = (size_t)(wsp - (char*)d_ws);
  const size_t rem  = (ws_size > used) ? (ws_size - used) : 0;
  __bf16* Achunk = (__bf16*)wsp;   // interp chunk buffer uses the remainder

  // Stage 1: normalization + downsize dense.
  downsize_kernel<<<N, 64, 0, stream>>>(signal, nmean, nvar, w_down, b_down,
                                        bnd_s, bnd_b, x0);

  float* xin = x0;
  float* xout = x1;
  int Cin = DOWN;
  for (int blk = 0; blk < 4; ++blk) {
    const int T = outDims[blk];
    const int K = 40 * Cin;
    const long long bkn = (long long)T * K;
    cvt_f32_bf16<<<(unsigned)((bkn + 255) / 256), 256, 0, stream>>>(kerns[blk], Bk, bkn);

    // Chunk rows through workspace (multiple of the 128-row WG tile).
    long long cr = (long long)(rem / ((size_t)K * sizeof(__bf16)));
    cr = (cr / 128) * 128;
    if (cr < 128) cr = 128;
    if (cr > N)   cr = N;
    const int chunk = (int)cr;

    for (int r0 = 0; r0 < N; r0 += chunk) {
      const int rows = (N - r0 < chunk) ? (N - r0) : chunk;
      const long long tot = (long long)rows * 40 * Cin;
      interp_gather<<<(unsigned)((tot + 255) / 256), 256, 0, stream>>>(
          xin, bc_idx, bc_w, Achunk, r0, tot, Cin);
      dim3 g((unsigned)(T / 32), (unsigned)((rows + 127) / 128));
      isc_conv_wmma<<<g, 256, 0, stream>>>(Achunk, Bk, cbias[blk], bns[blk],
                                           bnb[blk], xout, rows, r0, T, Cin);
    }
    float* t = xin; xin = xout; xout = t;
    Cin = T;
  }

  // Final dense: logits = x @ w_out + b_out.
  const long long xn = (long long)N * 384;
  cvt_f32_bf16<<<(unsigned)((xn + 255) / 256), 256, 0, stream>>>(xin, xa, xn);
  transpose_cvt<<<(unsigned)((xn + 255) / 256), 256, 0, stream>>>(w_out, Bt, 384, N, xn);
  dim3 g((unsigned)((N + 31) / 32), (unsigned)((N + 127) / 128));
  gemm_out_wmma<<<g, 256, 0, stream>>>(xa, Bt, b_out, out, N, N, 384);
}
